// YawNetVLAD_13280038880027
// MI455X (gfx1250) — matrix operand order repaired and checked by reading, not compiled, not run on hardware
//
#include <hip/hip_runtime.h>
#include <cstdint>
#include <cmath>

// ---------------------------------------------------------------------------
// CDNA5 (gfx1250, wave32) YawNetVLAD.  bf16 stored as ushort (explicit
// converts); __bf16 appears only as the WMMA operand vector type.
// This round: re-enable TDM weight-tile DMA (single-variable change vs the
// known-good round-4 source).
// ---------------------------------------------------------------------------

typedef unsigned short bfu;
typedef __attribute__((ext_vector_type(16))) __bf16 v16bf;
typedef __attribute__((ext_vector_type(8)))  float  v8f;
typedef __attribute__((ext_vector_type(4)))  unsigned int u32x4;
typedef __attribute__((ext_vector_type(8)))  int    i32x8;
typedef __attribute__((ext_vector_type(4)))  int    i32x4;

union V16U {
  v16bf v;
  bfu   u[16];
};

#define USE_TDM 1
#define NEG_INF (-3.402823466e38f)

__device__ __forceinline__ bfu f2b(float f) {
  union { float f; unsigned int u; } x;
  x.f = f;
  unsigned int r = x.u + 0x7FFFu + ((x.u >> 16) & 1u);  // round-nearest-even
  return (bfu)(r >> 16);
}
__device__ __forceinline__ float b2f(bfu b) {
  union { float f; unsigned int u; } x;
  x.u = ((unsigned int)b) << 16;
  return x.f;
}

// ---------------------------------------------------------------------------
// fp32 -> bf16(ushort) elementwise convert
// ---------------------------------------------------------------------------
__global__ __launch_bounds__(256) void f2bf_kernel(const float* __restrict__ in,
                                                   bfu* __restrict__ out, size_t n) {
  size_t i = (size_t)blockIdx.x * 256 + threadIdx.x;
  if (i < n) out[i] = f2b(in[i]);
}

// ---------------------------------------------------------------------------
// Pack conv weights (O, Cin, 5, 5) fp32 -> (25, O, Cin) bf16
// ---------------------------------------------------------------------------
__global__ __launch_bounds__(256) void packw_kernel(const float* __restrict__ w,
                                                    bfu* __restrict__ wp,
                                                    int O, int Ci) {
  size_t total = (size_t)O * Ci * 25;
  size_t i = (size_t)blockIdx.x * 256 + threadIdx.x;
  if (i >= total) return;
  int tap = (int)(i % 25);
  int c   = (int)((i / 25) % Ci);
  int o   = (int)(i / (25 * (size_t)Ci));
  wp[((size_t)tap * O + o) * Ci + c] = f2b(w[i]);
}

// ---------------------------------------------------------------------------
// Soft-assignment: logits = x . conv_w + b ; softmax over K=64 ; store bf16
// 64 threads = 64 spatial positions per block.
// ---------------------------------------------------------------------------
__global__ __launch_bounds__(64) void assign_softmax_kernel(
    const float* __restrict__ x, const float* __restrict__ cw,
    const float* __restrict__ cb, bfu* __restrict__ a) {
  __shared__ float wl[64 * 128];   // 32 KB
  __shared__ float bl[64];
  __shared__ bfu   xc[64][130];    // 16.6 KB, padded
  int t = threadIdx.x;
  for (int idx = t; idx < 64 * 128; idx += 64) wl[idx] = cw[idx];
  bl[t] = cb[t];
  size_t p = (size_t)blockIdx.x * 64 + t;  // total 172800 = 2700*64 exactly
  int n = (int)(p / (48 * 900));
  int rem = (int)(p % (48 * 900));
  int h = rem / 900, w = rem % 900;
  for (int c = 0; c < 128; ++c)
    xc[t][c] = f2b(x[(((size_t)n * 128 + c) * 48 + h) * 900 + w]);
  __syncthreads();
  float lg[64];
  for (int k = 0; k < 64; ++k) {
    float acc = bl[k];
    for (int c = 0; c < 128; ++c) acc += b2f(xc[t][c]) * wl[k * 128 + c];
    lg[k] = acc;
  }
  float m = lg[0];
  for (int k = 1; k < 64; ++k) m = fmaxf(m, lg[k]);
  float s = 0.f;
  for (int k = 0; k < 64; ++k) { lg[k] = __expf(lg[k] - m); s += lg[k]; }
  float inv = 1.f / s;
  for (int k = 0; k < 64; ++k)
    a[(((size_t)n * 64 + k) * 48 + h) * 900 + w] = f2b(lg[k] * inv);
}

// ---------------------------------------------------------------------------
// VLAD window GEMM: for (n, j): vlad_raw[k,c] = sum_{h, wi in win} a * x
// M=64 x N=128, K reduced over 48*200 window samples.  8 waves, 4 acc/wave.
// ---------------------------------------------------------------------------
__global__ __launch_bounds__(256) void vlad_gemm_kernel(
    const bfu* __restrict__ a, const bfu* __restrict__ x,
    float* __restrict__ vlad_raw) {
  __shared__ bfu a_tile[64][32];
  __shared__ bfu x_tile[128][32];
  int t = threadIdx.x, wave = t >> 5, lane = t & 31, l = lane & 15;
  int hi2 = (lane >= 16) ? 1 : 0;
  int n = blockIdx.x / 30, j = blockIdx.x % 30;
  int wb = (j * 30 + 800) % 900;  // circular window base
  int mt = wave >> 1, nq = wave & 1;
  v8f acc0 = {0.f,0.f,0.f,0.f,0.f,0.f,0.f,0.f};
  v8f acc1 = acc0, acc2 = acc0, acc3 = acc0;

#pragma unroll 1
  for (int h = 0; h < 48; ++h) {
#pragma unroll 1
    for (int s = 0; s < 7; ++s) {  // 7*32 = 224 >= 200 window cols
      __syncthreads();
      for (int idx = t; idx < (64 + 128) * 32; idx += 256) {
        int r = idx >> 5, wi = idx & 31;
        int wg = s * 32 + wi;
        bfu v = 0;
        if (wg < 200) {
          int w = wb + wg; if (w >= 900) w -= 900;
          if (r < 64)
            v = a[(((size_t)n * 64 + r) * 48 + h) * 900 + w];
          else
            v = x[(((size_t)n * 128 + (r - 64)) * 48 + h) * 900 + w];
        }
        if (r < 64) a_tile[r][wi] = v; else x_tile[r - 64][wi] = v;
      }
      __syncthreads();
      V16U af;
#pragma unroll
      for (int e = 0; e < 16; ++e) {
        int k = e + ((e >= 8) ? 8 : 0) + hi2 * 8;  // 16-bit A-operand layout
        af.u[e] = a_tile[mt * 16 + l][k];
      }
      V16U b0, b1, b2, b3;
#pragma unroll
      for (int e = 0; e < 16; ++e) {
        int ke = e + ((e >= 8) ? 8 : 0) + hi2 * 8;  // 16-bit B-operand layout
        b0.u[e] = x_tile[(nq * 4 + 0) * 16 + l][ke];
        b1.u[e] = x_tile[(nq * 4 + 1) * 16 + l][ke];
        b2.u[e] = x_tile[(nq * 4 + 2) * 16 + l][ke];
        b3.u[e] = x_tile[(nq * 4 + 3) * 16 + l][ke];
      }
      acc0 = __builtin_amdgcn_wmma_f32_16x16x32_bf16(false, af.v, false, b0.v,
                                                     (short)0, acc0, false, false);
      acc1 = __builtin_amdgcn_wmma_f32_16x16x32_bf16(false, af.v, false, b1.v,
                                                     (short)0, acc1, false, false);
      acc2 = __builtin_amdgcn_wmma_f32_16x16x32_bf16(false, af.v, false, b2.v,
                                                     (short)0, acc2, false, false);
      acc3 = __builtin_amdgcn_wmma_f32_16x16x32_bf16(false, af.v, false, b3.v,
                                                     (short)0, acc3, false, false);
    }
  }
  size_t base = (((size_t)n * 30 + j) * 64) * 128;
#pragma unroll
  for (int r = 0; r < 8; ++r) {
    int kk = mt * 16 + r + hi2 * 8;
    vlad_raw[base + (size_t)kk * 128 + (nq * 4 + 0) * 16 + l] = acc0[r];
    vlad_raw[base + (size_t)kk * 128 + (nq * 4 + 1) * 16 + l] = acc1[r];
    vlad_raw[base + (size_t)kk * 128 + (nq * 4 + 2) * 16 + l] = acc2[r];
    vlad_raw[base + (size_t)kk * 128 + (nq * 4 + 3) * 16 + l] = acc3[r];
  }
}

// ---------------------------------------------------------------------------
// asum[n,k,j] = sum over window (h,wi) of a
// ---------------------------------------------------------------------------
__global__ __launch_bounds__(256) void asum_kernel(const bfu* __restrict__ a,
                                                   float* __restrict__ asum) {
  int b = blockIdx.x;  // 4*64*30
  int n = b / (64 * 30);
  int rem = b % (64 * 30);
  int k = rem / 30, j = rem % 30;
  int wb = (j * 30 + 800) % 900;
  float s = 0.f;
#pragma unroll 1
  for (int idx = threadIdx.x; idx < 9600; idx += 256) {
    int h = idx / 200, wi = idx % 200;
    int w = wb + wi; if (w >= 900) w -= 900;
    s += b2f(a[(((size_t)n * 64 + k) * 48 + h) * 900 + w]);
  }
  __shared__ float red[256];
  red[threadIdx.x] = s;
  __syncthreads();
  for (int st = 128; st > 0; st >>= 1) {
    if (threadIdx.x < st) red[threadIdx.x] += red[threadIdx.x + st];
    __syncthreads();
  }
  if (threadIdx.x == 0) asum[((size_t)n * 64 + k) * 30 + j] = red[0];
}

// ---------------------------------------------------------------------------
// VLAD finalize: subtract centers*asum, intra-L2 over C, global L2 over 8192.
// ---------------------------------------------------------------------------
__global__ __launch_bounds__(256) void vlad_finalize_kernel(
    const float* __restrict__ vlad_raw, const float* __restrict__ centers,
    const float* __restrict__ asum, float* __restrict__ feat) {
  int n = blockIdx.x / 30, j = blockIdx.x % 30;
  __shared__ float v[64][128];
  __shared__ float rs[64][4];
  __shared__ float rscale[64];
  __shared__ float gred[64];
  __shared__ float gtotal;
  int t = threadIdx.x, k = t >> 2, q = t & 3;
  float as = asum[((size_t)n * 64 + k) * 30 + j];
  float pss = 0.f;
  for (int i = 0; i < 32; ++i) {
    int c = q * 32 + i;
    float val = vlad_raw[(((size_t)n * 30 + j) * 64 + k) * 128 + c]
              - centers[k * 128 + c] * as;
    v[k][c] = val;
    pss += val * val;
  }
  rs[k][q] = pss;
  __syncthreads();
  if (q == 0) {
    float ssum = rs[k][0] + rs[k][1] + rs[k][2] + rs[k][3];
    float sc = 1.f / fmaxf(sqrtf(ssum), 1e-12f);
    rscale[k] = sc;
    gred[k] = ssum * sc * sc;
  }
  __syncthreads();
  if (t == 0) {
    float s = 0.f;
    for (int i = 0; i < 64; ++i) s += gred[i];
    gtotal = 1.f / fmaxf(sqrtf(s), 1e-12f);
  }
  __syncthreads();
  float gs = gtotal, rsch = rscale[k];
  size_t row = (size_t)n * 30 + j;
  for (int i = 0; i < 32; ++i) {
    int c = q * 32 + i;
    feat[row * 9216 + k * 128 + c] = v[k][c] * rsch * gs;
  }
}

// ---------------------------------------------------------------------------
// TDM: DMA a 64-row x 32-col bf16 tile (row stride = stride_elems) into LDS.
// D# per CDNA5 ISA §8.  6-arg builtin form (frontend-verified round 1):
//   (u32x4 g0, i32x8 g1, i32x4 g2, i32x4 g3, i32x8 extra, i32 cpol)
// ---------------------------------------------------------------------------
__device__ __forceinline__ void tdm_load_w64x32(const bfu* gsrc, bfu* lds_dst,
                                                int stride_elems) {
#if USE_TDM
  unsigned long long ga = (unsigned long long)(uintptr_t)gsrc;
  unsigned int lofs = (unsigned int)(unsigned long long)(uintptr_t)lds_dst;
  u32x4 d0;
  d0[0] = 1u;                                            // count = 1 valid D#
  d0[1] = lofs;                                          // lds_addr
  d0[2] = (unsigned int)ga;                              // global_addr lo
  d0[3] = (unsigned int)((ga >> 32) & 0x01FFFFFFu) | (2u << 30);  // hi | type=2
  i32x8 d1;
  d1[0] = (int)(1u << 16);     // workgroup_mask=0, data_size=1 (2 bytes)
  d1[1] = (int)(32u << 16);    // tensor_dim0 lo16 = 32
  d1[2] = (int)(64u << 16);    // tensor_dim0 hi=0 | tensor_dim1 lo16 = 64
  d1[3] = (int)(32u << 16);    // tensor_dim1 hi=0 | tile_dim0 = 32
  d1[4] = (int)64;             // tile_dim1 = 64 | tile_dim2 = 0
  d1[5] = stride_elems;        // tensor_dim0_stride lo32
  d1[6] = 0;                   // stride0 hi | stride1 lo
  d1[7] = 0;                   // stride1 hi
  i32x4 dz = {0, 0, 0, 0};
  i32x8 dz8 = {0, 0, 0, 0, 0, 0, 0, 0};
  __builtin_amdgcn_tensor_load_to_lds(d0, d1, dz, dz, dz8, 0);
#else
  (void)gsrc; (void)lds_dst; (void)stride_elems;
#endif
}

// ---------------------------------------------------------------------------
// Implicit-GEMM 5x5 conv, pad=2.  A = input patch (16 pos x 32 cin),
// B = packed weights (32 cin x 16 outs).  Tile: 64 outs x 32 pos, 8 waves.
// circular=1 (layer1): logical col jj -> input col (jj+800)%900.
// ---------------------------------------------------------------------------
__global__ __launch_bounds__(256) void conv5x5_wmma_kernel(
    const bfu* __restrict__ in, const bfu* __restrict__ wpack,
    const float* __restrict__ bias, bfu* __restrict__ out,
    int Cin, int Cout, int Hh, int Wstore, int circular, int doRelu) {
  __shared__ bfu in_tile[32][40];   // [cin][36 cols + pad]
  __shared__ bfu w_tile[64 * 32];   // [o][cin]  (TDM destination)
  int t = threadIdx.x, wave = t >> 5, lane = t & 31, l = lane & 15;
  int hi2 = (lane >= 16) ? 1 : 0;
  int j0 = blockIdx.x * 32;
  int obase = blockIdx.y * 64;
  int n = blockIdx.z / Hh, i = blockIdx.z % Hh;
  int psub = wave & 1, osub = wave >> 1;

  v8f acc = {0.f, 0.f, 0.f, 0.f, 0.f, 0.f, 0.f, 0.f};

#pragma unroll 1
  for (int di = 0; di < 5; ++di) {
    int ii = i + di - 2;
    if (ii < 0 || ii >= Hh) continue;  // zero row contributes nothing
#pragma unroll 1
    for (int cb = 0; cb < Cin; cb += 32) {
      __syncthreads();
      // stage 32 x 36 patch (cols j0-2 .. j0+33) with zero / circular edges
      for (int idx = t; idx < 32 * 36; idx += 256) {
        int r = idx / 36, col = idx % 36;
        int jj = j0 - 2 + col;
        bfu v = 0;
        if (jj >= 0 && jj < 1100) {
          int js = circular ? ((jj + 800) % 900) : jj;
          v = in[(((size_t)n * Cin + (cb + r)) * Hh + ii) * Wstore + js];
        }
        in_tile[r][col] = v;
      }
#pragma unroll 1
      for (int dj = 0; dj < 5; ++dj) {
        const bfu* wsrc =
            wpack + (((size_t)(di * 5 + dj) * Cout + obase) * Cin + cb);
        __syncthreads();
#if USE_TDM
        if (wave == 0) tdm_load_w64x32(wsrc, w_tile, Cin);
        __builtin_amdgcn_s_wait_tensorcnt(0);
#else
        for (int idx = t; idx < 64 * 32; idx += 256) {
          int o = idx >> 5, c = idx & 31;
          w_tile[o * 32 + c] = wsrc[(size_t)o * Cin + c];
        }
#endif
        __syncthreads();
        V16U af, bm;
#pragma unroll
        for (int e = 0; e < 16; ++e) {
          int k = e + ((e >= 8) ? 8 : 0) + hi2 * 8;
          af.u[e] = in_tile[k][psub * 16 + l + dj];
        }
#pragma unroll
        for (int e = 0; e < 16; ++e)
          bm.u[e] = w_tile[(osub * 16 + l) * 32 + e + ((e >= 8) ? 8 : 0) + hi2 * 8];
        acc = __builtin_amdgcn_wmma_f32_16x16x32_bf16(
            false, af.v, false, bm.v, (short)0, acc, false, false);
      }
    }
  }
  int o = obase + osub * 16 + l;
  float b = bias[o];
#pragma unroll
  for (int r = 0; r < 8; ++r) {
    int pos = psub * 16 + r + hi2 * 8;
    int j = j0 + pos;
    if (j < 1100) {
      float v = acc[r] + b;
      if (doRelu) v = v > 0.f ? v : 0.2f * v;
      out[(((size_t)n * Cout + o) * Hh + i) * 1100 + j] = f2b(v);
    }
  }
}

// ---------------------------------------------------------------------------
// Generic maxpool: vertical kernel KH == vertical stride, horizontal stride 1
// with pad PW; width fixed 1100 in and out.
// ---------------------------------------------------------------------------
__global__ __launch_bounds__(256) void maxpool_kernel(
    const bfu* __restrict__ in, bfu* __restrict__ out,
    int C, int Hin, int Hout, int KH, int KW, int PW) {
  size_t total = (size_t)4 * C * Hout * 1100;
  size_t idx = (size_t)blockIdx.x * 256 + threadIdx.x;
  if (idx >= total) return;
  int j = (int)(idx % 1100);
  size_t r = idx / 1100;
  int oi = (int)(r % Hout); r /= Hout;
  int c = (int)(r % C);
  int n = (int)(r / C);
  float m = NEG_INF;
  for (int di = 0; di < KH; ++di) {
    int ii = oi * KH + di;
    const bfu* p = in + (((size_t)n * C + c) * Hin + ii) * 1100;
    for (int dj = 0; dj < KW; ++dj) {
      int jj = j - PW + dj;
      if (jj < 0 || jj >= 1100) continue;
      m = fmaxf(m, b2f(p[jj]));
    }
  }
  out[idx] = f2b(m);
}

// ---------------------------------------------------------------------------
// Final pool (4x200, stride 4x1) + sample every 30 cols + L2-norm over 1024,
// write feat[row, 8192:9216].  h3 (4,1024,4,1100) bf16.
// ---------------------------------------------------------------------------
__global__ __launch_bounds__(256) void pool3_norm_kernel(
    const bfu* __restrict__ h3, float* __restrict__ feat) {
  int n = blockIdx.x / 30, j = blockIdx.x % 30;
  int t = threadIdx.x;
  __shared__ float vals[1024];
  __shared__ float red[256];
  __shared__ float inv;
  float pss = 0.f;
  for (int ci = 0; ci < 4; ++ci) {
    int c = t + ci * 256;
    float m = NEG_INF;
    for (int r = 0; r < 4; ++r) {
      const bfu* p = h3 + (((size_t)n * 1024 + c) * 4 + r) * 1100 + j * 30;
#pragma unroll 4
      for (int w = 0; w < 200; ++w) m = fmaxf(m, b2f(p[w]));
    }
    vals[c] = m;
    pss += m * m;
  }
  red[t] = pss;
  __syncthreads();
  for (int st = 128; st > 0; st >>= 1) {
    if (t < st) red[t] += red[t + st];
    __syncthreads();
  }
  if (t == 0) inv = 1.f / fmaxf(sqrtf(red[0]), 1e-12f);
  __syncthreads();
  float iv = inv;
  size_t row = (size_t)n * 30 + j;
  for (int ci = 0; ci < 4; ++ci) {
    int c = t + ci * 256;
    feat[row * 9216 + 8192 + c] = vals[c] * iv;
  }
}

// ---------------------------------------------------------------------------
// MLP head: out[row,o] = feat[row].mlp_w[o] + b[o], then L2 over 256.
// ---------------------------------------------------------------------------
__global__ __launch_bounds__(256) void mlp_norm_kernel(
    const float* __restrict__ feat, const float* __restrict__ mw,
    const float* __restrict__ mb, float* __restrict__ out) {
  int row = blockIdx.x, t = threadIdx.x;
  __shared__ float fr[9216];
  __shared__ float red[256];
  __shared__ float inv;
  for (int i = t; i < 9216; i += 256) fr[i] = feat[(size_t)row * 9216 + i];
  __syncthreads();
  float acc = mb[t];
  const float* wp = mw + (size_t)t * 9216;
#pragma unroll 4
  for (int i = 0; i < 9216; ++i) acc += fr[i] * wp[i];
  red[t] = acc * acc;
  __syncthreads();
  for (int st = 128; st > 0; st >>= 1) {
    if (t < st) red[t] += red[t + st];
    __syncthreads();
  }
  if (t == 0) inv = 1.f / fmaxf(sqrtf(red[0]), 1e-12f);
  __syncthreads();
  out[(size_t)row * 256 + t] = acc * inv;
}

// ---------------------------------------------------------------------------
extern "C" void kernel_launch(void* const* d_in, const int* in_sizes, int n_in,
                              void* d_out, int out_size, void* d_ws,
                              size_t ws_size, hipStream_t stream) {
  const float* x  = (const float*)d_in[0];
  const float* cc = (const float*)d_in[1];
  const float* cw = (const float*)d_in[2];
  const float* cb = (const float*)d_in[3];
  const float* w1 = (const float*)d_in[4];
  const float* b1 = (const float*)d_in[5];
  const float* w2 = (const float*)d_in[6];
  const float* b2 = (const float*)d_in[7];
  const float* w3 = (const float*)d_in[8];
  const float* b3 = (const float*)d_in[9];
  const float* mw = (const float*)d_in[10];
  const float* mb = (const float*)d_in[11];
  float* out = (float*)d_out;

  char* p = (char*)d_ws;
  auto alloc = [&](size_t bytes) -> char* {
    char* r = p;
    p += (bytes + 255) & ~(size_t)255;
    return r;
  };
  const size_t NXE = (size_t)4 * 128 * 48 * 900;
  bfu* x_bf   = (bfu*)alloc(NXE * 2);
  bfu* a_bf   = (bfu*)alloc((size_t)4 * 64 * 48 * 900 * 2);
  bfu* w1p    = (bfu*)alloc((size_t)25 * 256 * 128 * 2);
  bfu* w2p    = (bfu*)alloc((size_t)25 * 512 * 256 * 2);
  bfu* w3p    = (bfu*)alloc((size_t)25 * 1024 * 512 * 2);
  float* asum = (float*)alloc((size_t)4 * 64 * 30 * 4);
  float* vraw = (float*)alloc((size_t)120 * 8192 * 4);
  float* feat = (float*)alloc((size_t)120 * 9216 * 4);
  bfu* bufA   = (bfu*)alloc((size_t)4 * 256 * 48 * 1100 * 2);  // h1 then h2
  bfu* bufB   = (bfu*)alloc((size_t)4 * 256 * 16 * 1100 * 2);  // p1 then h3
  bfu* p2     = (bfu*)alloc((size_t)4 * 512 * 4 * 1100 * 2);
  if ((size_t)(p - (char*)d_ws) > ws_size) return;

  // --- precision conversion / weight packing ---
  f2bf_kernel<<<(unsigned)((NXE + 255) / 256), 256, 0, stream>>>(x, x_bf, NXE);
  packw_kernel<<<(256 * 128 * 25 + 255) / 256, 256, 0, stream>>>(w1, w1p, 256, 128);
  packw_kernel<<<(512 * 256 * 25 + 255) / 256, 256, 0, stream>>>(w2, w2p, 512, 256);
  packw_kernel<<<(1024 * 512 * 25 + 255) / 256, 256, 0, stream>>>(w3, w3p, 1024, 512);

  // --- VLAD branch ---
  assign_softmax_kernel<<<2700, 64, 0, stream>>>(x, cw, cb, a_bf);
  vlad_gemm_kernel<<<120, 256, 0, stream>>>(a_bf, x_bf, vraw);
  asum_kernel<<<7680, 256, 0, stream>>>(a_bf, asum);
  vlad_finalize_kernel<<<120, 256, 0, stream>>>(vraw, cc, asum, feat);

  // --- conv branch ---
  {
    dim3 g(35, 4, 4 * 48);
    conv5x5_wmma_kernel<<<g, 256, 0, stream>>>(x_bf, w1p, b1, bufA,
                                               128, 256, 48, 900, 1, 1);
  }
  maxpool_kernel<<<70400, 256, 0, stream>>>(bufA, bufB, 256, 48, 16, 3, 3, 1);
  {
    dim3 g(35, 8, 4 * 16);
    conv5x5_wmma_kernel<<<g, 256, 0, stream>>>(bufB, w2p, b2, bufA,
                                               256, 512, 16, 1100, 0, 1);
  }
  maxpool_kernel<<<35200, 256, 0, stream>>>(bufA, p2, 512, 16, 4, 4, 5, 2);
  {
    dim3 g(35, 16, 4 * 4);
    conv5x5_wmma_kernel<<<g, 256, 0, stream>>>(p2, w3p, b3, bufB,
                                               512, 1024, 4, 1100, 0, 0);
  }
  pool3_norm_kernel<<<120, 256, 0, stream>>>(bufB, feat);

  // --- head ---
  mlp_norm_kernel<<<120, 256, 0, stream>>>(feat, mw, mb, out);
}
